// LongRangeFeaturizer_73057393705578
// MI455X (gfx1250) — compile-verified, matrix operand order checked by reading, not run on hardware
//
#include <hip/hip_runtime.h>
#include <hip/hip_bf16.h>
#include <math.h>

typedef __attribute__((ext_vector_type(16))) _Float16 v16h;
typedef __attribute__((ext_vector_type(8)))  _Float16 v8h;
typedef __attribute__((ext_vector_type(8)))  float    v8f;

// Problem constants (from the reference setup)
#define NS   16          // systems
#define NA   512         // atoms / system
#define ND   64          // channels
#define NE   16384       // edges / system
#define KTOT 4912        // 17^3 - 1 k-vectors
#define KPAD 4928        // padded to multiple of 32 (pad entries have G=0)
#define KB32 (KPAD / 32) // 154 k-blocks of 32

#define SELF_C   0.7978845608028654f   // sqrt(2/pi)/sigma, sigma=1
#define BG_C     3.14159265358979323f  // pi * sigma^2
#define KCUT2    39.478417604357434f   // (2*pi/LRWL)^2
#define INV_SQRT2 0.70710678118654752f

// Swizzled f16 tensors are stored in WMMA-B-fragment order:
//   [rowblock rb = row/32][coltile dt = col/16][lane L 0..31][16 halves]
// lane L = (col%16) + 16*h ; elements i   -> row rb*32 + h*8 + i
//                            elements i+8 -> row rb*32 + h*8 + 16 + i
// A producer C-fragment chunk (8 contiguous rows, one column) is exactly the
// 16B half-chunk at element offset e0 = (off&16)?8:0 of lane
// L = (col%16) + 16*((off>>3)&1), where off = (row0 % 32).

// ---------------------------------------------------------------------------
// 1) charges = features @ W^T + b   (WMMA f16 in / f32 out)
//    writes fp32 charges + f16 charges in B-fragment-swizzled order.
// ---------------------------------------------------------------------------
__global__ __launch_bounds__(32) void charges_wmma_kernel(
    const float* __restrict__ feat, const float* __restrict__ W,
    const float* __restrict__ bias, float* __restrict__ qF,
    _Float16* __restrict__ qHs)
{
  const int lane    = threadIdx.x;
  const int half_id = lane >> 4;
  const int r       = lane & 15;
  const int m0      = blockIdx.x * 16;    // rows of features (S*N)
  const int n0      = blockIdx.y * 16;    // output channel tile
  const int col     = n0 + r;

  v8f acc = {};
  #pragma unroll
  for (int kstep = 0; kstep < ND; kstep += 32) {
    const int kb = kstep + half_id * 8;   // per-ISA 16-bit A layout
    const float* arow = feat + (size_t)(m0 + r) * ND + kb;
    const float* brow = W    + (size_t)col      * ND + kb;   // B[k][n] = W[n][k]
    v16h a, b;
    #pragma unroll
    for (int i = 0; i < 8; ++i) {
      a[i]     = (_Float16)arow[i];
      a[i + 8] = (_Float16)arow[16 + i];
      b[i]     = (_Float16)brow[i];
      b[i + 8] = (_Float16)brow[16 + i];
    }
    acc = __builtin_amdgcn_wmma_f32_16x16x32_f16(false, a, false, b,
                                                 (short)0, acc, false, false);
  }
  const float bv = bias[col];
  v8h hq;
  #pragma unroll
  for (int r8 = 0; r8 < 8; ++r8) {
    const int row = m0 + r8 + 8 * half_id;      // C/D layout: M = r8 + 8*(lane/16)
    const float v = acc[r8] + bv;
    qF[(size_t)row * ND + col] = v;
    hq[r8] = (_Float16)v;                       // 8 contiguous rows of column `col`
  }
  // swizzled f16 store: one 16B chunk per lane
  const int off = (m0 & 16) + 8 * half_id;
  _Float16* dst = qHs
      + (((size_t)(m0 >> 5) * 4 + blockIdx.y) * 32 + (r + 16 * ((off >> 3) & 1))) * 16
      + ((off & 16) ? 8 : 0);
  *(v8h*)dst = hq;
}

// ---------------------------------------------------------------------------
// 2) per-system reciprocal lattice, k-vectors, G(k)/vol, |det(cell)|
// ---------------------------------------------------------------------------
__global__ __launch_bounds__(256) void kgrid_kernel(
    const float* __restrict__ cells, float* __restrict__ kvec,
    float* __restrict__ Gvol, float* __restrict__ vol)
{
  const int s = blockIdx.x;
  __shared__ float rec[9];
  __shared__ float vsh;
  if (threadIdx.x == 0) {
    const float* c = cells + (size_t)s * 9;
    const float a00=c[0],a01=c[1],a02=c[2],a10=c[3],a11=c[4],a12=c[5],
                a20=c[6],a21=c[7],a22=c[8];
    const float c00 = a11*a22 - a12*a21;
    const float c01 = a12*a20 - a10*a22;
    const float c02 = a10*a21 - a11*a20;
    const float det = a00*c00 + a01*c01 + a02*c02;
    const float inv = 1.0f / det;
    const float i00=c00*inv, i01=(a02*a21-a01*a22)*inv, i02=(a01*a12-a02*a11)*inv;
    const float i10=c01*inv, i11=(a00*a22-a02*a20)*inv, i12=(a02*a10-a00*a12)*inv;
    const float i20=c02*inv, i21=(a01*a20-a00*a21)*inv, i22=(a00*a11-a01*a10)*inv;
    const float tp = 6.28318530717958647692f;   // 2*pi ; recip = 2*pi*inv(cell)^T
    rec[0]=tp*i00; rec[1]=tp*i10; rec[2]=tp*i20;
    rec[3]=tp*i01; rec[4]=tp*i11; rec[5]=tp*i21;
    rec[6]=tp*i02; rec[7]=tp*i12; rec[8]=tp*i22;
    vsh = fabsf(det);
    vol[s] = vsh;
  }
  __syncthreads();
  const float v = vsh;
  for (int kk = threadIdx.x; kk < KPAD; kk += 256) {
    float kx = 0.f, ky = 0.f, kz = 0.f, g = 0.f;
    if (kk < KTOT) {
      const int idx = kk + (kk >= 2456 ? 1 : 0);   // skip (0,0,0) at flat idx 2456
      const int i = idx / 289       - 8;
      const int j = (idx / 17) % 17 - 8;
      const int l = idx % 17        - 8;
      kx = i*rec[0] + j*rec[3] + l*rec[6];
      ky = i*rec[1] + j*rec[4] + l*rec[7];
      kz = i*rec[2] + j*rec[5] + l*rec[8];
      const float ksq = kx*kx + ky*ky + kz*kz;
      if (ksq <= KCUT2)
        g = 12.566370614359172f / ksq * __expf(-0.5f * ksq);  // 4*pi/k^2 e^{-s^2k^2/2}
    }
    float* kv = kvec + ((size_t)s * KPAD + kk) * 3;
    kv[0] = kx; kv[1] = ky; kv[2] = kz;
    Gvol[(size_t)s * KPAD + kk] = g / v;          // fold 1/vol into G
  }
}

// ---------------------------------------------------------------------------
// 3) short-range edge scatter: pot_sr[i] += q[j] * sr(dist)
// ---------------------------------------------------------------------------
__global__ __launch_bounds__(256) void sr_kernel(
    const int* __restrict__ nidx, const float* __restrict__ ndist,
    const float* __restrict__ qF, float* __restrict__ potSR)
{
  const long long gid = (long long)blockIdx.x * 256 + threadIdx.x;
  const long long e   = gid >> 6;                 // global edge id
  const int d         = (int)(gid & 63);
  if (e >= (long long)NS * NE) return;
  const int s    = (int)(e / NE);
  const int i    = nidx[2 * e + 0];
  const int j    = nidx[2 * e + 1];
  const float dist = ndist[e];
  const float lr   = erff(dist * INV_SQRT2) / dist;
  const float fcut = (dist < 5.0f)
                   ? 0.5f * (1.0f + __cosf(0.62831853071795864769f * dist)) // pi*d/5
                   : 0.0f;
  const float sr = -lr * fcut;
  const float val = qF[((size_t)s * NA + j) * ND + d] * sr;
  atomicAdd(&potSR[((size_t)s * NA + i) * ND + d], val);
}

// ---------------------------------------------------------------------------
// 4) structure factors:  GSc/GSs[k,d] = (G/vol) * sum_n {cos|sin}(pos_n.k) q[n,d]
//    WMMA GEMM [K,N]x[N,D]; A generated via sincos; B loaded as swizzled
//    32B-per-lane fragments; output stored swizzled for the lr kernel.
// ---------------------------------------------------------------------------
__global__ __launch_bounds__(32) void sf_wmma_kernel(
    const float* __restrict__ pos, const _Float16* __restrict__ qHs,
    const float* __restrict__ kvec, const float* __restrict__ Gvol,
    _Float16* __restrict__ GScS, _Float16* __restrict__ GSsS)
{
  const int lane    = threadIdx.x;
  const int half_id = lane >> 4;
  const int r       = lane & 15;
  const int s       = blockIdx.z;
  const int k0      = blockIdx.x * 16;   // k-row tile
  const int dt      = blockIdx.y;        // channel tile
  const int col     = dt * 16 + r;

  const float* kv = kvec + ((size_t)s * KPAD + k0 + r) * 3;  // my A row's k-vector
  const float kx = kv[0], ky = kv[1], kz = kv[2];
  const float* ps = pos + (size_t)s * NA * 3;
  // swizzled charges: row-block stride = 4*512 halves; s starts at rb = s*16
  const _Float16* qB = qHs + (((size_t)s * 16) * 4 + dt) * 512 + (size_t)lane * 16;

  v8f accC = {}, accS = {};
  for (int nstep = 0; nstep < NA; nstep += 32) {
    const int nb = nstep + half_id * 8;
    v16h ac, as;
    #pragma unroll
    for (int i = 0; i < 8; ++i) {
      {
        const int n = nb + i;
        const float ph = ps[n*3+0]*kx + ps[n*3+1]*ky + ps[n*3+2]*kz;
        float sv, cv; __sincosf(ph, &sv, &cv);
        ac[i] = (_Float16)cv; as[i] = (_Float16)sv;
      }
      {
        const int n = nb + 16 + i;
        const float ph = ps[n*3+0]*kx + ps[n*3+1]*ky + ps[n*3+2]*kz;
        float sv, cv; __sincosf(ph, &sv, &cv);
        ac[i+8] = (_Float16)cv; as[i+8] = (_Float16)sv;
      }
    }
    const v16h b = *(const v16h*)(qB + (size_t)(nstep >> 5) * 2048); // 2x b128
    accC = __builtin_amdgcn_wmma_f32_16x16x32_f16(false, ac, false, b,
                                                  (short)0, accC, false, false);
    accS = __builtin_amdgcn_wmma_f32_16x16x32_f16(false, as, false, b,
                                                  (short)0, accS, false, false);
  }
  // scale by G/vol and store swizzled (one 16B chunk per lane per tensor)
  const int off = (k0 & 16) + 8 * half_id;
  const size_t base =
      (((size_t)s * KB32 + (k0 >> 5)) * 4 + dt) * 512
      + (size_t)(r + 16 * ((off >> 3) & 1)) * 16 + ((off & 16) ? 8 : 0);
  const float* gv = Gvol + (size_t)s * KPAD + k0 + 8 * half_id;
  v8h oc, os;
  #pragma unroll
  for (int r8 = 0; r8 < 8; ++r8) {
    const float g = gv[r8];                       // k = k0 + 8*half_id + r8
    oc[r8] = (_Float16)(accC[r8] * g);
    os[r8] = (_Float16)(accS[r8] * g);
  }
  *(v8h*)(GScS + base) = oc;
  *(v8h*)(GSsS + base) = os;
}

// ---------------------------------------------------------------------------
// 5) pot_lr[n,d] = sum_k cos(pos_n.k) GSc[k,d] + sin(pos_n.k) GSs[k,d]
//    4 waves/block share the B tiles via async global->LDS copies
//    (double-buffered, ASYNCcnt), fragments read with ds_load_b128.
// ---------------------------------------------------------------------------
#define LR_WPB 4
__global__ __launch_bounds__(32 * LR_WPB) void lr_wmma_kernel(
    const float* __restrict__ pos, const float* __restrict__ kvec,
    const _Float16* __restrict__ GScS, const _Float16* __restrict__ GSsS,
    float* __restrict__ potLR)
{
  __shared__ _Float16 smem[2 * 1024];   // 2 buffers x (GSc 512 + GSs 512 halves)
  const int tid     = threadIdx.x;
  const int wave    = tid >> 5;
  const int lane    = tid & 31;
  const int half_id = lane >> 4;
  const int r       = lane & 15;
  const int s       = blockIdx.z;
  const int n0      = (blockIdx.x * LR_WPB + wave) * 16;   // atom tile of this wave
  const int dt      = blockIdx.y;
  const int col     = dt * 16 + r;

  const float* p  = pos + ((size_t)s * NA + n0 + r) * 3;
  const float px = p[0], py = p[1], pz = p[2];
  const float* kvs = kvec + (size_t)s * KPAD * 3;

  // swizzled B: tile base for k-block rb is +rb*2048 halves
  const _Float16* gcB = GScS + ((size_t)s * KB32 * 4 + dt) * 512;
  const _Float16* gsB = GSsS + ((size_t)s * KB32 * 4 + dt) * 512;

  // cooperative async copy: thread t copies one 16B chunk; t<64 -> GSc, else GSs
  const int cpart = tid >> 6;
  const int chunk = tid & 63;
  const unsigned lds_base = (unsigned)(size_t)&smem[0];   // low 32 bits = LDS offset

  auto issue = [&](int st, int buf) {
    const _Float16* src = (cpart ? gsB : gcB) + (size_t)st * 2048 + (size_t)chunk * 8;
    const unsigned dst = lds_base + (unsigned)(buf * 2048 + cpart * 1024 + chunk * 16);
    const unsigned long long ga = (unsigned long long)(size_t)src;
    asm volatile("global_load_async_to_lds_b128 %0, %1, off"
                 :: "v"(dst), "v"(ga) : "memory");
  };

  issue(0, 0);
  v8f acc = {};
  for (int st = 0; st < KB32; ++st) {
    asm volatile("s_wait_asynccnt 0x0" ::: "memory");   // my chunk of buf[st&1] landed
    __syncthreads();                                    // everyone's chunks landed
    if (st + 1 < KB32) issue(st + 1, (st + 1) & 1);     // overlap next copy w/ compute

    const int kb = st * 32 + half_id * 8;
    v16h ac, as;
    #pragma unroll
    for (int i = 0; i < 8; ++i) {
      const int k1 = kb + i, k2 = kb + 16 + i;
      const float ph1 = kvs[k1*3+0]*px + kvs[k1*3+1]*py + kvs[k1*3+2]*pz;
      const float ph2 = kvs[k2*3+0]*px + kvs[k2*3+1]*py + kvs[k2*3+2]*pz;
      float s1, c1, s2, c2;
      __sincosf(ph1, &s1, &c1);
      __sincosf(ph2, &s2, &c2);
      ac[i] = (_Float16)c1;  ac[i+8] = (_Float16)c2;
      as[i] = (_Float16)s1;  as[i+8] = (_Float16)s2;
    }
    const int bb = (st & 1) * 1024;                     // halves
    const v16h bc = *(const v16h*)&smem[bb + lane * 16];        // 2x ds_load_b128
    const v16h bs = *(const v16h*)&smem[bb + 512 + lane * 16];
    acc = __builtin_amdgcn_wmma_f32_16x16x32_f16(false, ac, false, bc,
                                                 (short)0, acc, false, false);
    acc = __builtin_amdgcn_wmma_f32_16x16x32_f16(false, as, false, bs,
                                                 (short)0, acc, false, false);
  }
  #pragma unroll
  for (int r8 = 0; r8 < 8; ++r8) {
    const int row = n0 + r8 + 8 * half_id;
    potLR[((size_t)s * NA + row) * ND + col] = acc[r8];
  }
}

// ---------------------------------------------------------------------------
// 6) qsum[s,d] = sum_n q[s,n,d]
// ---------------------------------------------------------------------------
__global__ __launch_bounds__(64) void qsum_kernel(
    const float* __restrict__ qF, float* __restrict__ qsum)
{
  const int s = blockIdx.x, d = threadIdx.x;
  const float* p = qF + (size_t)s * NA * ND + d;
  float acc = 0.f;
  for (int n = 0; n < NA; ++n) acc += p[(size_t)n * ND];
  qsum[s * ND + d] = acc;
}

// ---------------------------------------------------------------------------
// 7) epilogue: out = (pot_sr + pot_lr - q*self_c - qsum*bg/vol) * q
// ---------------------------------------------------------------------------
__global__ __launch_bounds__(256) void final_kernel(
    const float* __restrict__ qF, const float* __restrict__ potSR,
    const float* __restrict__ potLR, const float* __restrict__ qsum,
    const float* __restrict__ vol, float* __restrict__ out)
{
  const size_t gid = (size_t)blockIdx.x * 256 + threadIdx.x;   // S*N*D elements
  const int d = (int)(gid & (ND - 1));
  const int s = (int)(gid >> 15);                              // / (NA*ND)
  const float q = qF[gid];
  const float corr = q * SELF_C + qsum[s * ND + d] * (BG_C / vol[s]);
  out[gid] = (potSR[gid] + potLR[gid] - corr) * q;
}

// ---------------------------------------------------------------------------
extern "C" void kernel_launch(void* const* d_in, const int* in_sizes, int n_in,
                              void* d_out, int out_size, void* d_ws, size_t ws_size,
                              hipStream_t stream) {
  (void)in_sizes; (void)n_in; (void)out_size; (void)ws_size;
  const float* feat  = (const float*)d_in[0];   // [S*N, D]
  const float* pos   = (const float*)d_in[1];   // [S, N, 3]
  const float* cells = (const float*)d_in[2];   // [S, 3, 3]
  const int*   nidx  = (const int*)  d_in[3];   // [S, E, 2]
  const float* ndist = (const float*)d_in[4];   // [S*E]
  const float* W     = (const float*)d_in[5];   // [D, D]
  const float* bias  = (const float*)d_in[6];   // [D]
  float* out = (float*)d_out;

  // workspace layout (all offsets 256B aligned)
  char* ws = (char*)d_ws;
  float*    qF    = (float*)   (ws + 0);          // S*N*D f32          2 MB
  _Float16* qHs   = (_Float16*)(ws + 2097152);    // swizzled f16       1 MB
  float*    potSR = (float*)   (ws + 3145728);    // S*N*D f32          2 MB
  float*    potLR = (float*)   (ws + 5242880);    // S*N*D f32          2 MB
  float*    kvec  = (float*)   (ws + 7340032);    // S*KPAD*3 f32       ~0.9 MB
  float*    Gvol  = (float*)   (ws + 8286208);    // S*KPAD f32         ~0.3 MB
  float*    vol   = (float*)   (ws + 8601600);    // S f32 (padded)
  _Float16* GScS  = (_Float16*)(ws + 8601856);    // swizzled f16       ~9.6 MB
  _Float16* GSsS  = (_Float16*)(ws + 18694400);   // swizzled f16       ~9.6 MB
  float*    qsum  = (float*)   (ws + 28786944);   // S*D f32

  // pot_sr accumulator must start at zero every call
  hipMemsetAsync(potSR, 0, (size_t)NS * NA * ND * sizeof(float), stream);

  // 1) charges GEMM (WMMA)
  charges_wmma_kernel<<<dim3(NS * NA / 16, ND / 16), 32, 0, stream>>>(
      feat, W, bias, qF, qHs);

  // 2) reciprocal lattice setup
  kgrid_kernel<<<NS, 256, 0, stream>>>(cells, kvec, Gvol, vol);

  // 3) short-range scatter
  sr_kernel<<<(NS * NE * ND) / 256, 256, 0, stream>>>(nidx, ndist, qF, potSR);

  // 4) structure factors (WMMA)
  sf_wmma_kernel<<<dim3(KPAD / 16, ND / 16, NS), 32, 0, stream>>>(
      pos, qHs, kvec, Gvol, GScS, GSsS);

  // 5) long-range potential (WMMA + async global->LDS double buffering)
  lr_wmma_kernel<<<dim3(NA / 16 / LR_WPB, ND / 16, NS), 32 * LR_WPB, 0, stream>>>(
      pos, kvec, GScS, GSsS, potLR);

  // 6) charge sums
  qsum_kernel<<<NS, ND, 0, stream>>>(qF, qsum);

  // 7) corrections + final multiply
  final_kernel<<<(NS * NA * ND) / 256, 256, 0, stream>>>(
      qF, potSR, potLR, qsum, vol, out);
}
